// Simple3PU_16638703305434
// MI455X (gfx1250) — matrix operand order repaired and checked by reading, not compiled, not run on hardware
//
#include <hip/hip_runtime.h>

// PU-Net-like pipeline on gfx1250: KNN -> EdgeConv (WMMA f16 GEMMs with fused
// training-mode BatchNorm) -> max-pool -> per-point decoder MLP. Two passes.

typedef __attribute__((ext_vector_type(16))) _Float16 v16h;
typedef __attribute__((ext_vector_type(8)))  float    v8f;

#define B_    16
#define N_    2048
#define KNN_  20
#define PN_   (B_ * N_)          // 32768 points
#define ROWS_ (PN_ * KNN_)       // 655360 edge rows

// ---------------------------------------------------------------- utilities
__global__ void k_zero(float* __restrict__ p, int n) {
    int i = blockIdx.x * 256 + threadIdx.x;
    if (i < n) p[i] = 0.0f;
}

// ---------------------------------------------------------------- KNN
// One thread per query point; candidates tiled through LDS; register
// insertion sort keeps top-20 by negative squared distance (self included,
// ties keep smaller index, matching lax.top_k on the reference's pd matrix).
__global__ void k_knn(const float* __restrict__ xyz, int* __restrict__ knn_out) {
    __shared__ float tx[256], ty[256], tz[256];
    const int b = blockIdx.y;
    const int q = blockIdx.x * 256 + threadIdx.x;
    const float* xb = xyz + (size_t)b * N_ * 3;
    const float qx = xb[q * 3 + 0], qy = xb[q * 3 + 1], qz = xb[q * 3 + 2];

    float best[KNN_];
    int   bidx[KNN_];
#pragma unroll
    for (int u = 0; u < KNN_; ++u) { best[u] = -3.4e38f; bidx[u] = 0; }

    for (int t = 0; t < N_; t += 256) {
        const int j = t + threadIdx.x;
        tx[threadIdx.x] = xb[j * 3 + 0];
        ty[threadIdx.x] = xb[j * 3 + 1];
        tz[threadIdx.x] = xb[j * 3 + 2];
        __syncthreads();
        for (int jj = 0; jj < 256; ++jj) {
            const float dx = qx - tx[jj], dy = qy - ty[jj], dz = qz - tz[jj];
            const float d = -(dx * dx + dy * dy + dz * dz);
            if (d > best[KNN_ - 1]) {
                float dd = d; int ii = t + jj;
#pragma unroll
                for (int u = 0; u < KNN_; ++u) {
                    if (dd > best[u]) {
                        float tf = best[u]; best[u] = dd; dd = tf;
                        int   ti = bidx[u]; bidx[u] = ii; ii = ti;
                    }
                }
            }
        }
        __syncthreads();
    }
    int* op = knn_out + (size_t)(b * N_ + q) * KNN_;
#pragma unroll
    for (int u = 0; u < KNN_; ++u) op[u] = bidx[u];
}

// ---------------------------------------------------------------- EdgeConv layer 1
// Build (nbr-ctr, ctr) 6-vector on the fly, 6->64 dense, write pre-BN h1,
// accumulate per-channel sum/sumsq for BatchNorm (training mode).
__global__ void k_edge1(const float* __restrict__ xyz, const int* __restrict__ knn_idx,
                        const float* __restrict__ w1, const float* __restrict__ b1,
                        float* __restrict__ h1, float* __restrict__ stats) {
    __shared__ float lw[64 * 6];
    __shared__ float lb[64];
    __shared__ float lsum[64], lsq[64];
    const int tid = threadIdx.x;
    for (int i = tid; i < 64 * 6; i += 256) lw[i] = w1[i];
    if (tid < 64) { lb[tid] = b1[tid]; lsum[tid] = 0.0f; lsq[tid] = 0.0f; }
    __syncthreads();

    const size_t r = (size_t)blockIdx.x * 256 + tid;
    const int b   = (int)(r / (N_ * KNN_));
    const int rem = (int)(r % (N_ * KNN_));
    const int n   = rem / KNN_;
    const float* xb = xyz + (size_t)b * N_ * 3;
    const float cx = xb[n * 3 + 0], cy = xb[n * 3 + 1], cz = xb[n * 3 + 2];
    const int nb = knn_idx[r];
    const float f[6] = { xb[nb * 3 + 0] - cx, xb[nb * 3 + 1] - cy, xb[nb * 3 + 2] - cz,
                         cx, cy, cz };
    float* hp = h1 + r * 64;
#pragma unroll
    for (int o = 0; o < 64; ++o) {
        float acc = lb[o];
#pragma unroll
        for (int c = 0; c < 6; ++c) acc += f[c] * lw[o * 6 + c];
        hp[o] = acc;
        atomicAdd(&lsum[o], acc);
        atomicAdd(&lsq[o], acc * acc);
    }
    __syncthreads();
    if (tid < 64) {
        atomicAdd(&stats[tid], lsum[tid]);
        atomicAdd(&stats[64 + tid], lsq[tid]);
    }
}

// ---------------------------------------------------------------- BN finalize
// s = g * rsqrt(var+eps), t = beta - mean*s  (y = x*s + t)
__global__ void k_bnfin(const float* __restrict__ stats, const float* __restrict__ g,
                        const float* __restrict__ be, float* __restrict__ st,
                        int C, float invCnt) {
    const int c = blockIdx.x * blockDim.x + threadIdx.x;
    if (c < C) {
        const float mean = stats[c] * invCnt;
        const float var  = stats[C + c] * invCnt - mean * mean;
        const float s = g[c] * rsqrtf(var + 1e-5f);
        st[c] = s;
        st[C + c] = be[c] - mean * s;
    }
}

// ---------------------------------------------------------------- weight fragment prep
// Swizzle W [Nc][K] (fp32, out-major) into the exact v_wmma_f32_16x16x32_f16
// B-operand layout: frag[((c*NT+nt)*32+lane)*16 + e] = W[(nt*16+lane%16)*K + 32c+8*(lane/16)+(e&7)+16*(e>>3)]
// (lane%16 = column within N-tile; lane/16 and element e give K within the 32-chunk).
__global__ void k_prep_wfrag(const float* __restrict__ W, _Float16* __restrict__ frag,
                             int K, int Nc) {
    int i = blockIdx.x * 256 + threadIdx.x;
    if (i >= K * Nc) return;
    const int e = i & 15;
    int rest = i >> 4;
    const int lane = rest & 31; rest >>= 5;
    const int NT = Nc >> 4;
    const int nt = rest % NT;
    const int c  = rest / NT;
    const int n = nt * 16 + (lane & 15);
    const int k = 32 * c + 8 * (lane >> 4) + (e & 7) + ((e >> 3) << 4);
    frag[i] = (_Float16)W[n * K + k];
}

// ---------------------------------------------------------------- WMMA GEMM
// C[M][N] = relu(A*s + t) @ Wfrag + bias, with per-channel sum/sumsq stats.
// Wave computes a 16xN stripe: A fragments (16-bit A layout: lanes 0-15 carry
// K 0-7/16-23, lanes 16-31 carry K 8-15/24-31 per 32-chunk) built from fp32
// with fused BN-affine+ReLU; B fragments are pre-swizzled fp16.
template <int K, int N>
__global__ void k_gemm_bnrelu_wmma(const float* __restrict__ A, const float* __restrict__ st,
                                   const _Float16* __restrict__ Bfrag,
                                   const float* __restrict__ bias,
                                   float* __restrict__ C, float* __restrict__ stats) {
    constexpr int KC = K / 32;
    constexpr int NT = N / 16;
    __shared__ float lsum[N], lsq[N];
    const int tid = threadIdx.x;
    for (int i = tid; i < N; i += 256) { lsum[i] = 0.0f; lsq[i] = 0.0f; }
    __syncthreads();

    const int wave = tid >> 5, lane = tid & 31;
    const int m = lane & 15, half = lane >> 4;
    const int rowBase = (blockIdx.x * 8 + wave) * 16;
    const float* rp = A + (size_t)(rowBase + m) * K;

    v16h a[KC];
#pragma unroll
    for (int c = 0; c < KC; ++c) {
        const int k0 = 32 * c + 8 * half;
        float xv[16], sv[16], tv[16];
        *(float4*)&xv[0]  = *(const float4*)(rp + k0);
        *(float4*)&xv[4]  = *(const float4*)(rp + k0 + 4);
        *(float4*)&xv[8]  = *(const float4*)(rp + k0 + 16);
        *(float4*)&xv[12] = *(const float4*)(rp + k0 + 20);
        *(float4*)&sv[0]  = *(const float4*)(st + k0);
        *(float4*)&sv[4]  = *(const float4*)(st + k0 + 4);
        *(float4*)&sv[8]  = *(const float4*)(st + k0 + 16);
        *(float4*)&sv[12] = *(const float4*)(st + k0 + 20);
        *(float4*)&tv[0]  = *(const float4*)(st + K + k0);
        *(float4*)&tv[4]  = *(const float4*)(st + K + k0 + 4);
        *(float4*)&tv[8]  = *(const float4*)(st + K + k0 + 16);
        *(float4*)&tv[12] = *(const float4*)(st + K + k0 + 20);
#pragma unroll
        for (int j = 0; j < 16; ++j)
            a[c][j] = (_Float16)fmaxf(xv[j] * sv[j] + tv[j], 0.0f);
    }

    const v16h* bf = (const v16h*)Bfrag;
#pragma unroll
    for (int nt = 0; nt < NT; ++nt) {
        v8f acc = {};
#pragma unroll
        for (int c = 0; c < KC; ++c) {
            const v16h bv = bf[(c * NT + nt) * 32 + lane];
            acc = __builtin_amdgcn_wmma_f32_16x16x32_f16(
                false, a[c], false, bv, (short)0, acc, false, false);
        }
        const int n = nt * 16 + m;           // lane -> column; half -> row group
        const float bs = bias[n];
        float ps = 0.0f, pq = 0.0f;
#pragma unroll
        for (int rr = 0; rr < 8; ++rr) {
            const float d = acc[rr] + bs;
            C[(size_t)(rowBase + rr + 8 * half) * N + n] = d;
            ps += d; pq += d * d;
        }
        ps += __shfl_xor(ps, 16);
        pq += __shfl_xor(pq, 16);
        if (half == 0) { atomicAdd(&lsum[n], ps); atomicAdd(&lsq[n], pq); }
    }
    __syncthreads();
    for (int i = tid; i < N; i += 256) {
        atomicAdd(&stats[i], lsum[i]);
        atomicAdd(&stats[N + i], lsq[i]);
    }
}

// ---------------------------------------------------------------- BN3 + max over k
__global__ void k_maxbn(const float* __restrict__ h3, const float* __restrict__ st,
                        float* __restrict__ feat) {
    const int i = blockIdx.x * 256 + threadIdx.x;  // over PN_*128
    const int c = i & 127, p = i >> 7;
    const float s = st[c], t = st[128 + c];
    const float* bp = h3 + (size_t)p * KNN_ * 128 + c;
    float m = -3.4e38f;
#pragma unroll
    for (int kk = 0; kk < KNN_; ++kk)
        m = fmaxf(m, bp[(size_t)kk * 128] * s + t);
    feat[i] = m;
}

// ---------------------------------------------------------------- decoder MLP
// [feat,(-0.2,-0.2)] 130 -> 64 -> 64 -> 3 per point; weights LDS-transposed.
__global__ void k_decoder(const float* __restrict__ feat,
                          const float* __restrict__ w1, const float* __restrict__ b1,
                          const float* __restrict__ w2, const float* __restrict__ b2,
                          const float* __restrict__ w3, const float* __restrict__ b3,
                          float* __restrict__ out) {
    __shared__ float lw1[130 * 64];   // [c][o]
    __shared__ float lw2[64 * 64];    // [c][o]
    __shared__ float lw3[64 * 3];     // [c][o]
    __shared__ float lb1[64], lb2[64], lb3[3];
    const int tid = threadIdx.x;
    for (int i = tid; i < 130 * 64; i += 128) { int o = i / 130, c = i % 130; lw1[c * 64 + o] = w1[i]; }
    for (int i = tid; i < 64 * 64;  i += 128) { int o = i / 64,  c = i % 64;  lw2[c * 64 + o] = w2[i]; }
    for (int i = tid; i < 3 * 64;   i += 128) { int o = i / 64,  c = i % 64;  lw3[c * 3 + o] = w3[i]; }
    if (tid < 64) { lb1[tid] = b1[tid]; lb2[tid] = b2[tid]; }
    if (tid < 3)  lb3[tid] = b3[tid];
    __syncthreads();

    const int p = blockIdx.x * 128 + tid;
    const float* fp = feat + (size_t)p * 128;
    float a1[64];
#pragma unroll
    for (int o = 0; o < 64; ++o) a1[o] = lb1[o];
    for (int c = 0; c < 128; ++c) {
        const float f = fp[c];
#pragma unroll
        for (int o = 0; o < 64; ++o) a1[o] += f * lw1[c * 64 + o];
    }
    for (int c = 128; c < 130; ++c) {
        const float f = -0.2f;
#pragma unroll
        for (int o = 0; o < 64; ++o) a1[o] += f * lw1[c * 64 + o];
    }
    float a2[64];
#pragma unroll
    for (int o = 0; o < 64; ++o) a2[o] = lb2[o];
    for (int c = 0; c < 64; ++c) {
        const float f = fmaxf(a1[c], 0.0f);
#pragma unroll
        for (int o = 0; o < 64; ++o) a2[o] += f * lw2[c * 64 + o];
    }
    float a3[3] = { lb3[0], lb3[1], lb3[2] };
    for (int c = 0; c < 64; ++c) {
        const float f = fmaxf(a2[c], 0.0f);
        a3[0] += f * lw3[c * 3 + 0];
        a3[1] += f * lw3[c * 3 + 1];
        a3[2] += f * lw3[c * 3 + 2];
    }
    out[p * 3 + 0] = a3[0];
    out[p * 3 + 1] = a3[1];
    out[p * 3 + 2] = a3[2];
}

// ---------------------------------------------------------------- launch
extern "C" void kernel_launch(void* const* d_in, const int* in_sizes, int n_in,
                              void* d_out, int out_size, void* d_ws, size_t ws_size,
                              hipStream_t stream) {
    (void)in_sizes; (void)n_in; (void)out_size; (void)ws_size;

    const float* xyz0 = (const float*)d_in[0];
    const float* ew1  = (const float*)d_in[1];
    const float* eb1  = (const float*)d_in[2];
    const float* eg1  = (const float*)d_in[3];
    const float* ebe1 = (const float*)d_in[4];
    const float* ew2  = (const float*)d_in[5];
    const float* eb2  = (const float*)d_in[6];
    const float* eg2  = (const float*)d_in[7];
    const float* ebe2 = (const float*)d_in[8];
    const float* ew3  = (const float*)d_in[9];
    const float* eb3  = (const float*)d_in[10];
    const float* eg3  = (const float*)d_in[11];
    const float* ebe3 = (const float*)d_in[12];

    char* w = (char*)d_ws;
    size_t off = 0;
    auto take = [&](size_t bytes) -> char* {
        char* p = w + off;
        off += (bytes + 255) & ~(size_t)255;
        return p;
    };
    int*       idxb   = (int*)take((size_t)ROWS_ * 4);
    float*     hA     = (float*)take((size_t)ROWS_ * 128 * 4);  // h1 (64ch) then h3 (128ch)
    float*     hB     = (float*)take((size_t)ROWS_ * 128 * 4);  // h2
    float*     feat   = (float*)take((size_t)PN_ * 128 * 4);
    float*     xyz1   = (float*)take((size_t)PN_ * 3 * 4);
    float*     stats  = (float*)take(640 * 4);                  // 128 + 256 + 256
    float*     st1    = (float*)take(128 * 4);
    float*     st2    = (float*)take(256 * 4);
    float*     st3    = (float*)take(256 * 4);
    _Float16*  w2frag = (_Float16*)take((size_t)64 * 128 * 2);
    _Float16*  w3frag = (_Float16*)take((size_t)128 * 128 * 2);

    float* stats1 = stats;
    float* stats2 = stats + 128;
    float* stats3 = stats + 384;

    k_prep_wfrag<<<(64 * 128 + 255) / 256, 256, 0, stream>>>(ew2, w2frag, 64, 128);
    k_prep_wfrag<<<(128 * 128 + 255) / 256, 256, 0, stream>>>(ew3, w3frag, 128, 128);

    const float invCnt = 1.0f / (float)ROWS_;
    for (int pass = 0; pass < 2; ++pass) {
        const float* xin = (pass == 0) ? xyz0 : xyz1;
        k_zero<<<(640 + 255) / 256, 256, 0, stream>>>(stats, 640);
        k_knn<<<dim3(N_ / 256, B_), 256, 0, stream>>>(xin, idxb);
        k_edge1<<<ROWS_ / 256, 256, 0, stream>>>(xin, idxb, ew1, eb1, hA, stats1);
        k_bnfin<<<1, 64, 0, stream>>>(stats1, eg1, ebe1, st1, 64, invCnt);
        k_gemm_bnrelu_wmma<64, 128><<<ROWS_ / 128, 256, 0, stream>>>(hA, st1, w2frag, eb2, hB, stats2);
        k_bnfin<<<1, 128, 0, stream>>>(stats2, eg2, ebe2, st2, 128, invCnt);
        k_gemm_bnrelu_wmma<128, 128><<<ROWS_ / 128, 256, 0, stream>>>(hB, st2, w3frag, eb3, hA, stats3);
        k_bnfin<<<1, 128, 0, stream>>>(stats3, eg3, ebe3, st3, 128, invCnt);
        k_maxbn<<<(PN_ * 128) / 256, 256, 0, stream>>>(hA, st3, feat);

        const float* dw1 = (const float*)d_in[13 + pass * 6 + 0];
        const float* db1 = (const float*)d_in[13 + pass * 6 + 1];
        const float* dw2 = (const float*)d_in[13 + pass * 6 + 2];
        const float* db2 = (const float*)d_in[13 + pass * 6 + 3];
        const float* dw3 = (const float*)d_in[13 + pass * 6 + 4];
        const float* db3 = (const float*)d_in[13 + pass * 6 + 5];
        float* outp = (pass == 0) ? xyz1 : (float*)d_out;
        k_decoder<<<PN_ / 128, 128, 0, stream>>>(feat, dw1, db1, dw2, db2, dw3, db3, outp);
    }
}
